// Generator_20435454394841
// MI455X (gfx1250) — matrix operand order repaired and checked by reading, not compiled, not run on hardware
//
#include <hip/hip_runtime.h>
#include <hip/hip_bf16.h>
#include <math.h>

typedef __bf16 bf16;
typedef __bf16 v16bf __attribute__((ext_vector_type(16)));
typedef float  v8f   __attribute__((ext_vector_type(8)));

#define B_    8
#define CIN   512
#define COUT  512
#define SDIM  512
#define H_    64
#define W_    64

#define LIN_SCALE  0.044194173824159216f   /* 1/sqrt(512)   */
#define CONV_SCALE 0.014731391274719739f   /* 1/sqrt(512*9) */

// Async global->LDS copy of 16 bytes per lane (CDNA5 TDM-lite path, ASYNCcnt).
__device__ __forceinline__ void async_copy16(const bf16* gsrc, bf16* ldst) {
  unsigned lds_off = (unsigned)(size_t)ldst;   // LDS aperture: addr[31:0] = LDS offset
  asm volatile("global_load_async_to_lds_b128 %0, %1, off"
               :: "v"(lds_off), "v"(gsrc) : "memory");
}
__device__ __forceinline__ void wait_async0() {
  asm volatile("s_wait_asynccnt 0" ::: "memory");
}

// ---------------- prep kernels ----------------

__global__ __launch_bounds__(256) void k_style(const float* __restrict__ style,
    const float* __restrict__ mw, const float* __restrict__ mb,
    float* __restrict__ s) {
  int i = blockIdx.x * 256 + threadIdx.x;          // b*512+ci
  if (i >= B_ * CIN) return;
  int b = i >> 9, ci = i & 511;
  const float* st = style + (size_t)b * SDIM;
  const float* w  = mw + (size_t)ci * SDIM;
  float acc = 0.f;
  for (int k = 0; k < SDIM; ++k) acc += st[k] * w[k];
  s[i] = acc * LIN_SCALE + mb[ci];
}

__global__ __launch_bounds__(256) void k_wsq(const float* __restrict__ cw,
                                             float* __restrict__ wsq) {
  int i = blockIdx.x * 256 + threadIdx.x;          // co*512+ci
  if (i >= COUT * CIN) return;
  const float* p = cw + (size_t)i * 9;
  float a = 0.f;
  for (int k = 0; k < 9; ++k) { float v = p[k]; a += v * v; }
  wsq[i] = a;
}

__global__ __launch_bounds__(256) void k_demod(const float* __restrict__ wsq,
    const float* __restrict__ s, float* __restrict__ demod) {
  int i = blockIdx.x * 256 + threadIdx.x;          // b*512+co
  if (i >= B_ * COUT) return;
  int b = i >> 9, co = i & 511;
  const float* wr = wsq + (size_t)co * CIN;
  const float* sr = s + (size_t)b * CIN;
  float a = 0.f;
  for (int ci = 0; ci < CIN; ++ci) { float sv = sr[ci]; a += wr[ci] * sv * sv; }
  demod[i] = rsqrtf(a * (CONV_SCALE * CONV_SCALE) + 1e-8f);
}

__global__ __launch_bounds__(256) void k_xscale(const float* __restrict__ x,
    const float* __restrict__ s, bf16* __restrict__ xb) {
  int i = blockIdx.x * 256 + threadIdx.x;          // b,ci,h,w flat (16777216)
  int ci = (i >> 12) & 511;
  int b  = i >> 21;
  xb[i] = (bf16)(x[i] * s[b * CIN + ci]);
}

__global__ __launch_bounds__(256) void k_wprep(const float* __restrict__ cw,
                                               bf16* __restrict__ wA) {
  int i = blockIdx.x * 256 + threadIdx.x;          // t*262144 + co*512 + ci
  int t = i / 262144;
  int rem = i & 262143;
  wA[i] = (bf16)(cw[(size_t)rem * 9 + t] * CONV_SCALE);
}

// ---------------- WMMA GEMM (per batch, per parity plane) ----------------
// out_p[co, y', x'] = demod[b,co] * sum_taps W_t[co,:] . x_s[:, y'-dy, x'-dx]
// Block tile: M=256 (Cout) x N=64 (pixels), K chunks of 32, double-buffered LDS.
// A tile staged with GLOBAL_LOAD_ASYNC_TO_LDS_B128 (no staging VGPRs).

__global__ __launch_bounds__(256, 1) void k_gemm(const bf16* __restrict__ xb,
    const bf16* __restrict__ wA, const float* __restrict__ demod,
    bf16* __restrict__ pre) {
  __shared__ __align__(16) bf16 Alds[2][256 * 32];  // 2 x 16 KB, [row(co)][k]
  __shared__ __align__(16) bf16 Blds[2][64 * 32];   // 2 x  4 KB, [n(pixel)][k]

  const int bp = blockIdx.z;
  const int b = bp >> 2, plane = bp & 3;
  const int Hp = (plane < 2) ? 65 : 64;
  const int Wp = (plane & 1) ? 64 : 65;
  const int Np = Hp * Wp;
  const int n_base = blockIdx.x * 64;
  if (n_base >= Np) return;                        // block-uniform exit
  const int co_block = blockIdx.y * 256;

  const int ntap_tbl[4]  = {4, 2, 2, 1};
  const int tapi_tbl[16] = {0,2,6,8,  1,7,0,0,  3,5,0,0,  4,0,0,0};
  const int tdy_tbl[16]  = {0,0,1,1,  0,1,0,0,  0,0,0,0,  0,0,0,0};
  const int tdx_tbl[16]  = {0,1,0,1,  0,0,0,0,  0,1,0,0,  0,0,0,0};
  const int nchunk = ntap_tbl[plane] * 16;         // (tap, ci0) flattened

  const size_t p1 = (size_t)B_ * COUT * 65 * 65;
  const size_t p2 = p1 + (size_t)B_ * COUT * 65 * 64;
  const size_t p3 = p2 + (size_t)B_ * COUT * 64 * 65;
  const size_t poff = (plane == 0) ? 0 : (plane == 1) ? p1 : (plane == 2) ? p2 : p3;

  const int tid  = threadIdx.x;
  const int lane = tid & 31;
  const int wave = tid >> 5;
  const int wm   = wave & 3;        // 64-row slab of M
  const int wn   = wave >> 2;       // 32-col slab of N
  const int lrow = lane & 15;
  const int half = lane >> 4;

  // ---- per-thread constant staging coordinates ----
  // A: 4 x 16B chunks/thread: row = it*64 + tid/4, col8 = (tid&3)*8
  const int arow0 = tid >> 2;
  const int acol8 = (tid & 3) * 8;
  // B: n = tid&63 is constant; kk = it*4 + tid/64, 8 elements/thread
  const int bn   = tid & 63;
  const int bkk0 = tid >> 6;
  const int ng0  = n_base + bn;
  const bool inplane = (ng0 < Np);
  const int yp0 = inplane ? (ng0 / Wp) : 0;
  const int xp0 = inplane ? (ng0 - yp0 * Wp) : 0;
  const size_t xbase = ((size_t)b * CIN + bkk0) * 4096;   // + ci0*4096 + it*4*4096

  bf16 breg[8];

  // A tile: issue async DMA global->LDS, no registers involved.
  auto load_A_async = [&](int c, int buf) {
    const int ti  = tapi_tbl[plane * 4 + (c >> 4)];
    const int ci0 = (c & 15) << 5;
    const bf16* wb = wA + (size_t)ti * COUT * CIN +
                     (size_t)(co_block + arow0) * CIN + ci0 + acol8;
    #pragma unroll
    for (int it = 0; it < 4; ++it)
      async_copy16(wb + (size_t)it * 64 * CIN,
                   &Alds[buf][(arow0 + it * 64) * 32 + acol8]);
  };

  // B tile: predicated gather into registers.
  auto load_B = [&](int c) {
    const int dy  = tdy_tbl[plane * 4 + (c >> 4)];
    const int dx  = tdx_tbl[plane * 4 + (c >> 4)];
    const int ci0 = (c & 15) << 5;
    const int ys = yp0 - dy, xs = xp0 - dx;
    const bool ok = inplane && ((unsigned)ys < 64u) && ((unsigned)xs < 64u);
    const size_t src = xbase + (size_t)ci0 * 4096 + ys * 64 + xs;
    #pragma unroll
    for (int it = 0; it < 8; ++it)
      breg[it] = ok ? xb[src + (size_t)it * 4 * 4096] : (bf16)0.f;
  };

  auto store_B = [&](int buf) {
    #pragma unroll
    for (int it = 0; it < 8; ++it)
      Blds[buf][bn * 32 + bkk0 + it * 4] = breg[it];
  };

  v8f acc[4][2];
  #pragma unroll
  for (int s2 = 0; s2 < 4; ++s2)
    #pragma unroll
    for (int nt = 0; nt < 2; ++nt)
      #pragma unroll
      for (int r = 0; r < 8; ++r) acc[s2][nt][r] = 0.f;

  auto compute = [&](int cur) {
    // Hold the two B fragments (16 VGPRs) live; stream A fragments through
    // one 8-VGPR temp, each used for both N subtiles immediately.
    union Frag { v16bf v; uint4 q[2]; };
    Frag bfr[2];
    #pragma unroll
    for (int nt = 0; nt < 2; ++nt) {
      const int ncol = wn * 32 + nt * 16 + lrow;
      bfr[nt].q[0] = *(const uint4*)&Blds[cur][ncol * 32 + half * 8];
      bfr[nt].q[1] = *(const uint4*)&Blds[cur][ncol * 32 + 16 + half * 8];
    }
    #pragma unroll
    for (int sub = 0; sub < 4; ++sub) {
      Frag afr;
      const int ar = wm * 64 + sub * 16 + lrow;
      afr.q[0] = *(const uint4*)&Alds[cur][ar * 32 + half * 8];
      afr.q[1] = *(const uint4*)&Alds[cur][ar * 32 + 16 + half * 8];
      #pragma unroll
      for (int nt = 0; nt < 2; ++nt)
        acc[sub][nt] = __builtin_amdgcn_wmma_f32_16x16x32_bf16(
            false, afr.v, false, bfr[nt].v, (short)0, acc[sub][nt], false, false);
    }
  };

  // ---- software-pipelined main loop: 1 barrier per chunk ----
  load_A_async(0, 0);
  load_B(0);
  store_B(0);
  wait_async0();
  __syncthreads();
  for (int c = 0; c < nchunk; ++c) {
    const int cur = c & 1;
    if (c + 1 < nchunk) {
      load_A_async(c + 1, cur ^ 1);                // DMA overlaps WMMA below
      load_B(c + 1);
    }
    compute(cur);
    if (c + 1 < nchunk) store_B(cur ^ 1);
    wait_async0();                                 // own async copies done
    __syncthreads();                               // everyone's copies visible
  }

  // ---- epilogue: scale by demod, store bf16 parity planes ----
  #pragma unroll
  for (int nt = 0; nt < 2; ++nt) {
    const int ng = n_base + wn * 32 + nt * 16 + lrow;
    if (ng < Np) {
      const int yp = ng / Wp, xp = ng - yp * Wp;
      #pragma unroll
      for (int sub = 0; sub < 4; ++sub) {
        #pragma unroll
        for (int r = 0; r < 8; ++r) {
          const int co = co_block + wm * 64 + sub * 16 + half * 8 + r;
          const float dm = demod[b * COUT + co];
          pre[poff + (((size_t)b * COUT + co) * Hp + yp) * Wp + xp] =
              (bf16)(acc[sub][nt][r] * dm);
        }
      }
    }
  }
}

// ---------------- blur + noise + bias + leaky ----------------

__global__ __launch_bounds__(256) void k_blur(const bf16* __restrict__ pre,
    const float* __restrict__ noise, const float* __restrict__ nw,
    const float* __restrict__ act_bias, float* __restrict__ out) {
  int idx = blockIdx.x * 256 + threadIdx.x;        // b,co,Y,X flat (67108864)
  int X  = idx & 127;
  int Y  = (idx >> 7) & 127;
  int co = (idx >> 14) & 511;
  int b  = idx >> 23;

  const size_t p1 = (size_t)B_ * COUT * 65 * 65;
  const size_t p2 = p1 + (size_t)B_ * COUT * 65 * 64;
  const size_t p3 = p2 + (size_t)B_ * COUT * 64 * 65;
  const size_t poff_[4] = {0, p1, p2, p3};
  const int Hp_[4] = {65, 65, 64, 64};
  const int Wp_[4] = {65, 64, 65, 64};
  const float bw[4] = {1.f, 3.f, 3.f, 1.f};

  float a = 0.f;
  #pragma unroll
  for (int dy = 0; dy < 4; ++dy) {
    int oy = Y - 1 + dy;
    if ((unsigned)oy >= 129u) continue;
    #pragma unroll
    for (int dx = 0; dx < 4; ++dx) {
      int ox = X - 1 + dx;
      if ((unsigned)ox >= 129u) continue;
      int p  = ((oy & 1) << 1) | (ox & 1);
      int yp = oy >> 1, xp = ox >> 1;
      float v = (float)pre[poff_[p] + (((size_t)b * COUT + co) * Hp_[p] + yp) * Wp_[p] + xp];
      a += v * (bw[dy] * bw[dx] * (1.f / 16.f));   // outer/64 * up^2(=4)
    }
  }
  a += nw[0] * noise[((size_t)b * 128 + Y) * 128 + X];
  a += act_bias[co];
  a = (a >= 0.f ? a : 0.2f * a) * 1.4142135623730951f;
  out[idx] = a;
}

// ---------------- launch ----------------

extern "C" void kernel_launch(void* const* d_in, const int* in_sizes, int n_in,
                              void* d_out, int out_size, void* d_ws, size_t ws_size,
                              hipStream_t stream) {
  const float* x     = (const float*)d_in[0];
  const float* style = (const float*)d_in[1];
  const float* noise = (const float*)d_in[2];
  const float* cw    = (const float*)d_in[3];
  const float* mw    = (const float*)d_in[4];
  const float* mb    = (const float*)d_in[5];
  const float* nw    = (const float*)d_in[6];
  const float* ab    = (const float*)d_in[7];
  float* out = (float*)d_out;

  char* ws = (char*)d_ws;
  float* s     = (float*)(ws + 0);            //  16 KB
  float* demod = (float*)(ws + 16384);        //  16 KB
  float* wsq   = (float*)(ws + 32768);        //   1 MB
  bf16*  wA    = (bf16*)(ws + 1081344);       // 4.5 MB : 9 taps x 512x512
  bf16*  xb    = (bf16*)(ws + 5799936);       //  32 MB : scaled x, bf16
  bf16*  pre   = (bf16*)(ws + 39354368);      // 130 MB : 4 parity planes

  k_style <<<16,     256, 0, stream>>>(style, mw, mb, s);
  k_wsq   <<<1024,   256, 0, stream>>>(cw, wsq);
  k_demod <<<16,     256, 0, stream>>>(wsq, s, demod);
  k_xscale<<<65536,  256, 0, stream>>>(x, s, xb);
  k_wprep <<<9216,   256, 0, stream>>>(cw, wA);

  dim3 g(67, 2, 32);                          // N tiles x M tiles x (batch*plane)
  k_gemm  <<<g,      256, 0, stream>>>(xb, wA, demod, pre);

  k_blur  <<<262144, 256, 0, stream>>>(pre, noise, nw, ab, out);
}
